// SparseMoELayer_63393717289150
// MI455X (gfx1250) — compile-verified, matrix operand order checked by reading, not run on hardware
//
#include <hip/hip_runtime.h>
#include <hip/hip_bf16.h>
#include <math.h>

// ---------------------------------------------------------------------------
// Problem constants (from reference): B=8, S=4096, D=256, F=512, E=8, K=2
// ---------------------------------------------------------------------------
#define B_  8
#define S_  4096
#define D_  256
#define F_  512
#define E_  8
#define TM  64          // tokens per block in the fused FFN kernel
#define FC  64          // F-chunk size
#define NCHUNK 16       // 2 experts * 8 F-chunks

typedef __bf16 bf16_t;
typedef __attribute__((ext_vector_type(16))) __bf16 bf16x16;
typedef __attribute__((ext_vector_type(8)))  float  floatx8;
typedef __attribute__((ext_vector_type(4)))  int    v4i_t;

#define AS1 __attribute__((address_space(1)))
#define AS3 __attribute__((address_space(3)))

// LDS layout (bytes) for the fused kernel (double-buffered weight staging)
#define SA_OFF     0                           // x tile  bf16 [64][264]
#define SA_BYTES   (64*264*2)                  // 33792
#define SW1_ELEMS  (64*264)                    // one W1^T chunk buffer (bf16)
#define SW1_OFF    (SA_OFF + SA_BYTES)         // 2 buffers: 67584 B
#define SW2_ELEMS  (256*72)                    // one W2^T chunk buffer (bf16)
#define SW2_OFF    (SW1_OFF + 2*SW1_ELEMS*2)   // 2 buffers: 73728 B
#define LNB_OFF    SW1_OFF                     // overlay: f32 [64][260] = 66560 B
#define SHC_OFF    (SW2_OFF + 2*SW2_ELEMS*2)   // h chunk bf16 [64][72] = 9216 B
#define CB2_OFF    (SHC_OFF + 64*72*2)         // combined b2, f32 [256]
#define SMEM_BYTES (CB2_OFF + 256*4)           // 185344 (< 320KB/WGP)

// ---------------------------------------------------------------------------
// Async global->LDS copy (CDNA5 GLOBAL_LOAD_ASYNC_TO_LDS_B128, ASYNCcnt) with
// compile-probed fallback to a synchronous copy.
// Builtin signature (from hipcc diagnostic): (v4i AS1*, v4i AS3*, imm, imm).
// ---------------------------------------------------------------------------
#if __has_builtin(__builtin_amdgcn_global_load_async_to_lds_b128)
#define HAVE_ASYNC_LDS 1
__device__ __forceinline__ void cp16_g2l(const bf16_t* g, bf16_t* l) {
    __builtin_amdgcn_global_load_async_to_lds_b128(
        (AS1 v4i_t*)(g), (AS3 v4i_t*)(l), 0, 0);
}
#else
#define HAVE_ASYNC_LDS 0
__device__ __forceinline__ void cp16_g2l(const bf16_t* g, bf16_t* l) {
    *(uint4*)l = *(const uint4*)g;
}
#endif

#if HAVE_ASYNC_LDS
#if __has_builtin(__builtin_amdgcn_s_wait_asynccnt)
#define WAIT_ASYNC(n) __builtin_amdgcn_s_wait_asynccnt(n)
#else
#define WAIT_ASYNC(n) asm volatile("s_wait_asynccnt %0" ::"i"(n) : "memory")
#endif
#else
#define WAIT_ASYNC(n)
#endif

// Scheduling fence: keep the ds_load fragment batch before the WMMA batch so
// the register allocator cannot coalesce fragments into one register quad
// (which would serialize each WMMA behind an s_wait_dscnt 0 LDS round-trip).
#if __has_builtin(__builtin_amdgcn_sched_barrier)
#define SCHED_FENCE() __builtin_amdgcn_sched_barrier(0)
#else
#define SCHED_FENCE()
#endif

// Load one 16x32 bf16 A/B fragment from LDS (row-major, 16B-aligned rows).
// Lane L<16 holds row (L), K {kb..kb+7, kb+16..kb+23}; L>=16 adds kb+=8.
__device__ __forceinline__ bf16x16 ld_frag(const bf16_t* p) {
    union { uint4 u[2]; bf16x16 v; } un;
    un.u[0] = *(const uint4*)(p);
    un.u[1] = *(const uint4*)(p + 16);
    return un.v;
}

// ---------------------------------------------------------------------------
// Kernel 1: partial sums over tokens for the per-batch mean.  grid (32, B)
// ---------------------------------------------------------------------------
__global__ __launch_bounds__(256) void k_partial_mean(const float* __restrict__ x,
                                                      float* __restrict__ partial) {
    const int b = blockIdx.y, c = blockIdx.x, tid = threadIdx.x;  // tid == d
    const float* p = x + ((size_t)b * S_ + (size_t)c * 128) * D_ + tid;
    float s = 0.f;
    for (int i = 0; i < 128; ++i) s += p[(size_t)i * D_];
    partial[((size_t)b * 32 + c) * D_ + tid] = s;
}

// ---------------------------------------------------------------------------
// Kernel 2: router (single block). mean -> logits -> softmax -> capacity ->
// softmax -> per-batch top-2 -> top-2 softmax.  Outputs eidx[B][2], ew[B][2].
// ---------------------------------------------------------------------------
__global__ __launch_bounds__(256) void k_router(const float* __restrict__ partial,
                                                const float* __restrict__ Wr,
                                                const float* __restrict__ br,
                                                int* __restrict__ eidx,
                                                float* __restrict__ ew) {
    __shared__ float xm[B_][D_];
    __shared__ float lg[B_][E_];
    __shared__ float rw[B_][E_];
    __shared__ float scale[E_];
    const int tid = threadIdx.x;

    for (int b = 0; b < B_; ++b) {
        float s = 0.f;
        for (int c = 0; c < 32; ++c) s += partial[((size_t)b * 32 + c) * D_ + tid];
        xm[b][tid] = s * (1.0f / (float)S_);
    }
    __syncthreads();
    if (tid < B_ * E_) {
        const int b = tid >> 3, e = tid & 7;
        float l = br[e];
        for (int d = 0; d < D_; ++d) l += xm[b][d] * Wr[d * E_ + e];
        lg[b][e] = l;
    }
    __syncthreads();
    if (tid < B_) {                       // first softmax
        const int b = tid;
        float mx = lg[b][0];
        for (int e = 1; e < E_; ++e) mx = fmaxf(mx, lg[b][e]);
        float s = 0.f;
        for (int e = 0; e < E_; ++e) { float v = __expf(lg[b][e] - mx); rw[b][e] = v; s += v; }
        for (int e = 0; e < E_; ++e) rw[b][e] /= s;
    }
    __syncthreads();
    if (tid < E_) {                       // capacity: ceil(N*1.25/E) = 5120
        const int e = tid;
        float load = 0.f;
        for (int b = 0; b < B_; ++b) load += rw[b][e];
        load *= (float)S_;                // * (N/B)
        scale[e] = (load > 5120.0f) ? (5120.0f / load) : 1.0f;
    }
    __syncthreads();
    if (tid < B_) {
        const int b = tid;
        float t[E_];
        float mx = -1e30f;
        for (int e = 0; e < E_; ++e) { t[e] = rw[b][e] * scale[e]; mx = fmaxf(mx, t[e]); }
        float s = 0.f;
        for (int e = 0; e < E_; ++e) { t[e] = __expf(t[e] - mx); s += t[e]; }
        for (int e = 0; e < E_; ++e) t[e] /= s;
        int e0 = 0;
        for (int e = 1; e < E_; ++e) if (t[e] > t[e0]) e0 = e;     // strict > : lowest-index ties
        int e1 = (e0 == 0) ? 1 : 0;
        for (int e = 0; e < E_; ++e) if (e != e0 && t[e] > t[e1]) e1 = e;
        const float a = __expf(t[e0]), c = __expf(t[e1]);          // softmax over top-2 values
        const float w0 = a / (a + c);
        eidx[2 * b] = e0; eidx[2 * b + 1] = e1;
        ew[2 * b] = w0;   ew[2 * b + 1] = 1.0f - w0;
    }
}

// ---------------------------------------------------------------------------
// Kernel 3: convert + transpose weights to bf16.
// W1T[e][f][d] = bf16(W1[e][d][f]);  W2T[e][d][f] = bf16(W2[e][f][d])
// ---------------------------------------------------------------------------
__global__ __launch_bounds__(256) void k_convert_w(const float* __restrict__ W1,
                                                   const float* __restrict__ W2,
                                                   bf16_t* __restrict__ W1T,
                                                   bf16_t* __restrict__ W2T) {
    const size_t n = (size_t)E_ * D_ * F_;   // 1048576
    size_t gid = (size_t)blockIdx.x * 256 + threadIdx.x;
    if (gid < n) {
        const size_t o = gid;
        const int d = (int)(o & 255), f = (int)((o >> 8) & 511), e = (int)(o >> 17);
        W1T[o] = (bf16_t)W1[((size_t)e * D_ + d) * F_ + f];
    } else {
        const size_t o = gid - n;
        const int f = (int)(o & 511), d = (int)((o >> 9) & 255), e = (int)(o >> 17);
        W2T[o] = (bf16_t)W2[((size_t)e * F_ + f) * D_ + d];
    }
}

// ---------------------------------------------------------------------------
// Kernel 4: fused top-2 expert FFN + combine + residual + LayerNorm.
// grid (S/TM, B), 256 threads = 8 wave32s.  Tile: 64 tokens x full D=256.
// Weight chunks double-buffered via async global->LDS copies (ASYNCcnt).
// ---------------------------------------------------------------------------
__global__ __launch_bounds__(256) void k_moe_ffn(const float* __restrict__ x,
                                                 const float* __restrict__ b1,
                                                 const float* __restrict__ b2,
                                                 const float* __restrict__ gamma,
                                                 const float* __restrict__ beta,
                                                 const bf16_t* __restrict__ W1T,
                                                 const bf16_t* __restrict__ W2T,
                                                 const int* __restrict__ eidx,
                                                 const float* __restrict__ ew,
                                                 float* __restrict__ out) {
    extern __shared__ __align__(16) unsigned char smem[];
    bf16_t* sA   = (bf16_t*)(smem + SA_OFF);
    bf16_t* sW1b = (bf16_t*)(smem + SW1_OFF);   // 2 buffers of SW1_ELEMS
    bf16_t* sW2b = (bf16_t*)(smem + SW2_OFF);   // 2 buffers of SW2_ELEMS
    bf16_t* sHC  = (bf16_t*)(smem + SHC_OFF);
    float*  lnb  = (float*)(smem + LNB_OFF);
    float*  cb2  = (float*)(smem + CB2_OFF);

    const int tid  = threadIdx.x;
    const int lane = tid & 31;
    const int w    = tid >> 5;
    const int b    = blockIdx.y;
    const int row0 = blockIdx.x * TM;

    const int e0  = eidx[2 * b], e1 = eidx[2 * b + 1];
    const float wk0 = ew[2 * b], wk1 = ew[2 * b + 1];

    // combined output bias: w0*b2[e0] + w1*b2[e1]
    cb2[tid] = wk0 * b2[e0 * D_ + tid] + wk1 * b2[e1 * D_ + tid];

    // stage x tile (fp32 -> bf16) into LDS, row-major [64][264]
    const float* xblk = x + ((size_t)b * S_ + row0) * D_;
    for (int i = 0; i < 16; ++i) {
        const int l = tid + i * 256;        // float4 units, 4096 total
        const int r = l >> 6, c4 = l & 63;
        float4 v = ((const float4*)(xblk + (size_t)r * D_))[c4];
        bf16_t* d = sA + r * 264 + c4 * 4;
        d[0] = (bf16_t)v.x; d[1] = (bf16_t)v.y; d[2] = (bf16_t)v.z; d[3] = (bf16_t)v.w;
    }

    // Issue one (expert, F-chunk) weight stage: 16 async b128 copies / thread.
    auto issue_chunk = [&](int c) {
        const int kq = c >> 3, fc = c & 7;
        const int ee = kq ? e1 : e0;
        const bf16_t* w1src = W1T + (size_t)ee * F_ * D_ + (size_t)fc * FC * D_;
        const bf16_t* w2src = W2T + (size_t)ee * D_ * F_ + fc * FC;
        bf16_t* dW1 = sW1b + (size_t)(c & 1) * SW1_ELEMS;
        bf16_t* dW2 = sW2b + (size_t)(c & 1) * SW2_ELEMS;
        for (int i = 0; i < 8; ++i) {       // W1^T chunk: rows f(64) x d(256)
            const int l = tid + i * 256;    // 2048 b128 units
            const int r = l >> 5, cc = l & 31;
            cp16_g2l(w1src + (size_t)r * D_ + cc * 8, dW1 + r * 264 + cc * 8);
        }
        for (int i = 0; i < 8; ++i) {       // W2^T chunk: rows d(256) x f(64)
            const int l = tid + i * 256;
            const int r = l >> 3, cc = l & 7;
            cp16_g2l(w2src + (size_t)r * F_ + cc * 8, dW2 + r * 72 + cc * 8);
        }
    };

    floatx8 oacc[8] = {};                   // 64x256 output, 8 tiles per wave
    const int mt  = w & 3;                  // row-tile 0..3
    const int nb1 = (w >> 2) * 2;           // gemm1 col-tile base (0 or 2)
    const int ntb = (w >> 2) * 8;           // gemm2 col-tile base (0 or 8)
    const int kb_off = (lane & 16) ? 8 : 0;
    const int arow   = mt * 16 + (lane & 15);
    const int rbase  = mt * 16 + ((lane & 16) ? 8 : 0);

    issue_chunk(0);                          // prologue

    for (int c = 0; c < NCHUNK; ++c) {
        const int   kq = c >> 3, fc = c & 7;
        const int   e  = kq ? e1 : e0;
        const float wk = kq ? wk1 : wk0;
        const bf16_t* bW1 = sW1b + (size_t)(c & 1) * SW1_ELEMS;
        const bf16_t* bW2 = sW2b + (size_t)(c & 1) * SW2_ELEMS;

        if (c + 1 < NCHUNK) {
            issue_chunk(c + 1);              // overlap next chunk's staging
            WAIT_ASYNC(16);                  // in-order: chunk c has landed
        } else {
            WAIT_ASYNC(0);
        }
        __syncthreads();                     // chunk c visible to all waves

        // ---- GEMM1: h[64xFC] = x_tile @ W1chunk (K = D = 256) ----
        floatx8 hacc[2] = {};
        for (int kk = 0; kk < D_; kk += 32) {
            bf16x16 a  = ld_frag(sA + arow * 264 + kk + kb_off);
            bf16x16 f0 = ld_frag(bW1 + (nb1 * 16 + (lane & 15)) * 264 + kk + kb_off);
            bf16x16 f1 = ld_frag(bW1 + ((nb1 + 1) * 16 + (lane & 15)) * 264 + kk + kb_off);
            SCHED_FENCE();                   // keep both fragments live -> no per-WMMA wait 0
            hacc[0] = __builtin_amdgcn_wmma_f32_16x16x32_bf16(
                false, a, false, f0, (short)0, hacc[0], false, false);
            hacc[1] = __builtin_amdgcn_wmma_f32_16x16x32_bf16(
                false, a, false, f1, (short)0, hacc[1], false, false);
        }
        // +b1, exact GELU, pre-scale by expert weight, -> bf16 LDS
        for (int t = 0; t < 2; ++t) {
            const int fl     = (nb1 + t) * 16 + (lane & 15);
            const float bias = b1[e * F_ + fc * FC + fl];
            for (int v = 0; v < 8; ++v) {
                float h = hacc[t][v] + bias;
                h = 0.5f * h * (1.0f + erff(h * 0.7071067811865475f));
                sHC[(rbase + v) * 72 + fl] = (bf16_t)(h * wk);
            }
        }
        __syncthreads();

        // ---- GEMM2 accumulate: out += (wk*h) @ W2chunk (K = FC = 64) ----
        // Batch all B-fragments, fence, then 8 back-to-back WMMAs.  The fence
        // forces distinct fragment registers so the waitcnt pass emits
        // incremental s_wait_dscnt instead of a full wait per tile.
        for (int kk = 0; kk < FC; kk += 32) {
            bf16x16 a = ld_frag(sHC + arow * 72 + kk + kb_off);
            bf16x16 bfr[8];
            for (int t = 0; t < 8; ++t)
                bfr[t] = ld_frag(bW2 + ((ntb + t) * 16 + (lane & 15)) * 72 + kk + kb_off);
            SCHED_FENCE();
            for (int t = 0; t < 8; ++t)
                oacc[t] = __builtin_amdgcn_wmma_f32_16x16x32_bf16(
                    false, a, false, bfr[t], (short)0, oacc[t], false, false);
        }
        __syncthreads();   // all reads of buffer (c&1) done before re-issue at c+2
    }

    // epilogue: + combined b2 + residual(x fp32) -> LDS (overlays weight bufs)
    for (int t = 0; t < 8; ++t) {
        const int d = (ntb + t) * 16 + (lane & 15);
        const float cv = cb2[d];
        for (int v = 0; v < 8; ++v) {
            const int m = rbase + v;
            lnb[m * 260 + d] = oacc[t][v] + cv + xblk[(size_t)m * D_ + d];
        }
    }
    __syncthreads();
    // LayerNorm: wave w handles rows w*8 .. w*8+7 (256 elems via 32 lanes)
    for (int r = w * 8; r < w * 8 + 8; ++r) {
        float vals[8], s = 0.f, s2 = 0.f;
        for (int i = 0; i < 8; ++i) {
            const float v = lnb[r * 260 + lane + i * 32];
            vals[i] = v; s += v; s2 += v * v;
        }
        for (int off = 16; off > 0; off >>= 1) {
            s  += __shfl_xor(s,  off, 32);
            s2 += __shfl_xor(s2, off, 32);
        }
        const float mu  = s * (1.0f / 256.0f);
        const float var = s2 * (1.0f / 256.0f) - mu * mu;
        const float rs  = rsqrtf(var + 1e-5f);
        float* orow = out + ((size_t)b * S_ + row0 + r) * D_;
        for (int i = 0; i < 8; ++i) {
            const int d = lane + i * 32;
            orow[d] = (vals[i] - mu) * rs * gamma[d] + beta[d];
        }
    }
}

// ---------------------------------------------------------------------------
extern "C" void kernel_launch(void* const* d_in, const int* in_sizes, int n_in,
                              void* d_out, int out_size, void* d_ws, size_t ws_size,
                              hipStream_t stream) {
    const float* x     = (const float*)d_in[0];
    const float* Wr    = (const float*)d_in[1];
    const float* br    = (const float*)d_in[2];
    const float* W1    = (const float*)d_in[3];
    const float* b1    = (const float*)d_in[4];
    const float* W2    = (const float*)d_in[5];
    const float* b2    = (const float*)d_in[6];
    const float* gamma = (const float*)d_in[7];
    const float* beta  = (const float*)d_in[8];
    float* out = (float*)d_out;

    char* ws = (char*)d_ws;
    float*  partial = (float*)ws;                                  // 8*32*256*4 = 262144 B
    int*    eidx    = (int*)(ws + 262144);                         // 64 B
    float*  ew      = (float*)(ws + 262144 + 64);                  // 64 B
    bf16_t* W1T     = (bf16_t*)(ws + 262144 + 128);                // 2 MB
    bf16_t* W2T     = (bf16_t*)(ws + 262144 + 128 + 2097152);      // 2 MB  (total ~4.5 MB)

    k_partial_mean<<<dim3(32, B_), 256, 0, stream>>>(x, partial);
    k_router<<<1, 256, 0, stream>>>(partial, Wr, br, eidx, ew);
    k_convert_w<<<(2u * E_ * D_ * F_) / 256, 256, 0, stream>>>(W1, W2, W1T, W2T);
    k_moe_ffn<<<dim3(S_ / TM, B_), 256, SMEM_BYTES, stream>>>(
        x, b1, b2, gamma, beta, W1T, W2T, eidx, ew, out);
}
